// CBOW_81913616270225
// MI455X (gfx1250) — matrix operand order repaired and checked by reading, not compiled, not run on hardware
//
#include <hip/hip_runtime.h>

// ---------- problem constants (from the reference) ----------
#define VOCAB 100000
#define CTX   8
#define EMB   64
#define BATCH 2048
#define KDIM  (CTX * EMB)   // 512

// ---------- GEMM tiling ----------
#define MT 128              // M tile per workgroup
#define NT 128              // N tile per workgroup
#define KC 64               // K chunk staged through LDS
#define LROW (KC + 8)       // 72 halves/row -> 144B stride, bank-conflict-free

typedef _Float16 half4 __attribute__((ext_vector_type(4)));
typedef _Float16 half8 __attribute__((ext_vector_type(8)));
typedef _Float16 v16h  __attribute__((ext_vector_type(16)));
typedef float    v8f   __attribute__((ext_vector_type(8)));
typedef int      v4i   __attribute__((ext_vector_type(4)));

#if defined(__gfx1250__) && __has_builtin(__builtin_amdgcn_global_load_async_to_lds_b128)
#define HAVE_ASYNC_LDS 1
#define AS1 __attribute__((address_space(1)))
#define AS3 __attribute__((address_space(3)))
#endif

// Phase 1: embedding gather  A[b*512 + c*64 + e] = emb_W[e, ctx[b,c]] + emb_b[e]  (f16)
__global__ __launch_bounds__(256)
void cbow_embed(const int* __restrict__ ctx, const float* __restrict__ embW,
                const float* __restrict__ embB, _Float16* __restrict__ A) {
    int t = blockIdx.x * blockDim.x + threadIdx.x;
    if (t >= BATCH * KDIM) return;
    int e  = t & (EMB - 1);
    int bc = t >> 6;                       // b*CTX + c
    int idx = ctx[bc];
    A[t] = (_Float16)(embW[(size_t)e * VOCAB + idx] + embB[e]);
}

// Phase 2: out[m,n] = sum_k A[m,k] * W[n,k] + bias[n]
// A: [2048 x 512] f16 (ws), W: [100000 x 512] f32, out: [2048 x 100000] f32
__global__ __launch_bounds__(256)
void cbow_gemm(const _Float16* __restrict__ A, const float* __restrict__ W,
               const float* __restrict__ bias, float* __restrict__ out) {
    __shared__ _Float16 As[MT * LROW];
    __shared__ _Float16 Ws[NT * LROW];

    const int tid   = threadIdx.x;
    const int wid   = tid >> 5;
    const int lane  = tid & 31;
    const int lhalf = lane >> 4;     // 0: lanes 0-15, 1: lanes 16-31
    const int lrow  = lane & 15;

    const int m0 = blockIdx.x * MT;  // M-tiles on x (fast) -> W strip reuse in L2
    const int n0 = blockIdx.y * NT;

    // 8 waves as 2(M) x 4(N); each wave owns a 64x32 subtile = 4x2 WMMA tiles
    const int wm = (wid >> 2) * 64;
    const int wn = (wid & 3)  * 32;

    // accumulators, bias folded in (C/D layout: lane&15 = N, vgpr v + 8*lhalf = M)
    v8f acc[4][2];
    #pragma unroll
    for (int nt = 0; nt < 2; ++nt) {
        int n = n0 + wn + nt * 16 + lrow;
        float bv = (n < VOCAB) ? bias[n] : 0.0f;
        #pragma unroll
        for (int mt = 0; mt < 4; ++mt)
            #pragma unroll
            for (int i = 0; i < 8; ++i) acc[mt][nt][i] = bv;
    }

    for (int k0 = 0; k0 < KDIM; k0 += KC) {
        // --- stage A tile: 128 rows x 64 halves, 16B chunks (1024 total, 4/thread)
#ifdef HAVE_ASYNC_LDS
        #pragma unroll
        for (int i = 0; i < 4; ++i) {
            int c   = tid + i * 256;
            int row = c >> 3;               // 8 chunks of 8 halves per row
            int col = (c & 7) * 8;
            v4i AS1* g = (v4i AS1*)(A + (size_t)(m0 + row) * KDIM + k0 + col);
            v4i AS3* l = (v4i AS3*)&As[row * LROW + col];
            __builtin_amdgcn_global_load_async_to_lds_b128(g, l, 0, 0);
        }
#else
        #pragma unroll
        for (int i = 0; i < 4; ++i) {
            int c   = tid + i * 256;
            int row = c >> 3;
            int col = (c & 7) * 8;
            half8 v = *(const half8*)(A + (size_t)(m0 + row) * KDIM + k0 + col);
            *(half8*)&As[row * LROW + col] = v;
        }
#endif
        // --- stage W tile: 128 rows x 64 f32 -> f16, float4 chunks (2048, 8/thread)
        #pragma unroll
        for (int i = 0; i < 8; ++i) {
            int c   = tid + i * 256;
            int row = c >> 4;               // 16 float4 per row
            int col = (c & 15) * 4;
            int n   = n0 + row;
            float4 v = {0.0f, 0.0f, 0.0f, 0.0f};
            if (n < VOCAB) v = *(const float4*)(W + (size_t)n * KDIM + k0 + col);
            half4 h;
            h[0] = (_Float16)v.x; h[1] = (_Float16)v.y;
            h[2] = (_Float16)v.z; h[3] = (_Float16)v.w;
            *(half4*)&Ws[row * LROW + col] = h;
        }
#ifdef HAVE_ASYNC_LDS
#if __has_builtin(__builtin_amdgcn_s_wait_asynccnt)
        __builtin_amdgcn_s_wait_asynccnt(0);
#else
        asm volatile("s_wait_asynccnt 0" ::: "memory");
#endif
#endif
        __syncthreads();

        #pragma unroll
        for (int ks = 0; ks < KC; ks += 32) {
            // A fragments: 16-bit A 16x32 layout
            //   lane<16 : K ks+0..7  and ks+16..23 of row M=lane
            //   lane>=16: K ks+8..15 and ks+24..31 of row M=lane-16
            v16h af[4];
            #pragma unroll
            for (int mt = 0; mt < 4; ++mt) {
                const _Float16* p = &As[(wm + mt * 16 + lrow) * LROW + ks + 8 * lhalf];
                half8 lo = *(const half8*)p;
                half8 hi = *(const half8*)(p + 16);
                af[mt] = __builtin_shufflevector(lo, hi,
                         0,1,2,3,4,5,6,7,8,9,10,11,12,13,14,15);
            }
            // B fragments: column n = lane&15; lane<16 holds K ks+0..7 and ks+16..23, lane>=16 K ks+8..15 and ks+24..31
            v16h bf[2];
            #pragma unroll
            for (int nt = 0; nt < 2; ++nt) {
                const _Float16* p = &Ws[(wn + nt * 16 + lrow) * LROW + ks + 8 * lhalf];
                half8 lo = *(const half8*)p;
                half8 hi = *(const half8*)(p + 16);
                bf[nt] = __builtin_shufflevector(lo, hi,
                         0,1,2,3,4,5,6,7,8,9,10,11,12,13,14,15);
            }
            #pragma unroll
            for (int mt = 0; mt < 4; ++mt)
                #pragma unroll
                for (int nt = 0; nt < 2; ++nt)
                    acc[mt][nt] = __builtin_amdgcn_wmma_f32_16x16x32_f16(
                        false, af[mt], false, bf[nt],
                        (short)0, acc[mt][nt], false, false);
        }
        __syncthreads();
    }

    // store (non-temporal: keep fc_W resident in L2; output is a one-shot 819MB stream)
    // D layout -> row m = v + 8*lhalf, col n = lane&15
    #pragma unroll
    for (int mt = 0; mt < 4; ++mt)
        #pragma unroll
        for (int nt = 0; nt < 2; ++nt) {
            int n = n0 + wn + nt * 16 + lrow;
            if (n < VOCAB) {
                #pragma unroll
                for (int v = 0; v < 8; ++v) {
                    int m = m0 + wm + mt * 16 + v + 8 * lhalf;
                    __builtin_nontemporal_store(acc[mt][nt][v],
                                                &out[(size_t)m * VOCAB + n]);
                }
            }
        }
}

extern "C" void kernel_launch(void* const* d_in, const int* in_sizes, int n_in,
                              void* d_out, int out_size, void* d_ws, size_t ws_size,
                              hipStream_t stream) {
    const int*   ctx  = (const int*)  d_in[0];   // contexts [2048, 8] int32
    const float* embW = (const float*)d_in[1];   // [64, 100000]
    const float* embB = (const float*)d_in[2];   // [64]
    const float* fcW  = (const float*)d_in[3];   // [100000, 512]
    const float* fcB  = (const float*)d_in[4];   // [100000]
    float* out = (float*)d_out;                  // [2048, 100000]
    _Float16* Ah = (_Float16*)d_ws;              // [2048, 512] f16 (2 MB scratch)

    cbow_embed<<<(BATCH * KDIM + 255) / 256, 256, 0, stream>>>(ctx, embW, embB, Ah);

    dim3 grid(BATCH / MT, (VOCAB + NT - 1) / NT);   // (16, 782)
    cbow_gemm<<<grid, 256, 0, stream>>>(Ah, fcW, fcB, out);
}
